// SelfAttention_5592047419596
// MI455X (gfx1250) — matrix operand order repaired and checked by reading, not compiled, not run on hardware
//
#include <hip/hip_runtime.h>

// Problem constants (match reference)
#define B_DIM 2
#define S_DIM 2048
#define DM    1024
#define H_DIM 16
#define DH    64
#define BH    (B_DIM * H_DIM)                       // 32
#define PROJ_ELEMS (B_DIM * H_DIM * S_DIM * DH)     // 4,194,304 per tensor

typedef __bf16 bf16;
typedef __bf16 v16bf  __attribute__((ext_vector_type(16)));
typedef __bf16 bf16x8 __attribute__((ext_vector_type(8)));
typedef __bf16 bf16x4 __attribute__((ext_vector_type(4)));
typedef float  v8f    __attribute__((ext_vector_type(8)));
typedef float  f32x4  __attribute__((ext_vector_type(4)));

// D = A(16x32 bf16) * B(32x16 bf16) + C(16x16 f32)
__device__ __forceinline__ v8f wmma_bf16(v16bf a, v16bf b, v8f c) {
  return __builtin_amdgcn_wmma_f32_16x16x32_bf16(false, a, false, b, (short)0, c,
                                                 false, false);
}

// gfx1250 async DMA: global memory -> LDS, tracked by ASYNCcnt.
// lds  = LDS byte address (low 32 bits of the flat shared pointer)
// gptr = 64-bit global address
__device__ __forceinline__ void async_copy_b128(unsigned int lds,
                                                const void* gptr) {
  asm volatile("global_load_async_to_lds_b128 %0, %1, off"
               :
               : "v"(lds), "v"(gptr)
               : "memory");
}
__device__ __forceinline__ void wait_asynccnt0() {
  asm volatile("s_wait_asynccnt 0x0" ::: "memory");
}
__device__ __forceinline__ unsigned int lds_off(const void* p) {
  return (unsigned int)(unsigned long long)p;   // LDS aperture: low 32 bits
}

// A-matrix fragment, 16x32, row-major LDS tile [m][k], stride ld elements.
// ISA layout: lane l (0..15) holds row M=l; VGPR v holds K pair at
// (v>>2)*16 + hi*8 + (v&3)*2  (hi = lane>=16).  -> two 16B LDS reads/lane.
__device__ __forceinline__ v16bf frag_a(const bf16* p, int ld, int lane) {
  const int l = lane & 15, hi = (lane >> 4) & 1;
  const bf16* row = p + l * ld;
  v16bf a;
#pragma unroll
  for (int v = 0; v < 8; ++v) {
    const int kk = ((v >> 2) << 4) + (hi << 3) + ((v & 3) << 1);
    a[2 * v]     = row[kk];
    a[2 * v + 1] = row[kk + 1];
  }
  return a;
}

// B-matrix fragment, 32x16, row-major LDS tile [k][n], stride ld elements.
// ISA layout: lane holds column N = l; element j holds K =
// (j>>3)*16 + hi*8 + (j&7)  (same K order as frag_a).
__device__ __forceinline__ v16bf frag_b(const bf16* p, int ld, int lane) {
  const int l = lane & 15, hi = (lane >> 4) & 1;
  const bf16* col = p + l;
  v16bf b;
#pragma unroll
  for (int j = 0; j < 16; ++j) b[j] = col[(((j >> 3) << 4) + (hi << 3) + (j & 7)) * ld];
  return b;
}

// ---------------------------------------------------------------------------
// Kernel 1: fused QKV projection GEMM.  grid = (N/64, M/128, 3), block = 256.
// Software-pipelined: tile k+1 global loads overlap tile k WMMAs.
// z selects (X, W, bias, Out).  K written transposed per head: [bh][d][s].
// ---------------------------------------------------------------------------
__global__ __launch_bounds__(256) void qkv_proj_kernel(
    const float* __restrict__ Xq, const float* __restrict__ Xk,
    const float* __restrict__ Xv, const float* __restrict__ Wq,
    const float* __restrict__ Wk, const float* __restrict__ Wv,
    const float* __restrict__ Bq, const float* __restrict__ Bk,
    const float* __restrict__ Bv, bf16* __restrict__ Qh,
    bf16* __restrict__ Kh, bf16* __restrict__ Vh) {
  constexpr int LDA = 40;   // 128x32 tile, +8 pad (80B rows, 8B aligned)
  constexpr int LDB = 72;   // 32x64 tile, +8 pad (144B rows, 16B aligned)
  __shared__ bf16 Al[128 * LDA];
  __shared__ bf16 Bl[32 * LDB];

  const int z = blockIdx.z;
  const float* X  = (z == 0) ? Xq : (z == 1) ? Xk : Xv;
  const float* W  = (z == 0) ? Wq : (z == 1) ? Wk : Wv;
  const float* Bs = (z == 0) ? Bq : (z == 1) ? Bk : Bv;
  bf16* Out       = (z == 0) ? Qh : (z == 1) ? Kh : Vh;

  const int tid  = threadIdx.x;
  const int wave = tid >> 5, lane = tid & 31;
  const int l = lane & 15, hi = (lane >> 4) & 1;
  const int m0 = blockIdx.y * 128;
  const int n0 = blockIdx.x * 64;

  // Per-thread staging geometry (compile-time trip counts, no divergence).
  int aRow[4], aCol[4], bRow[2], bCol[2];
#pragma unroll
  for (int j = 0; j < 4; ++j) {
    const int c = tid + j * 256;
    aRow[j] = c >> 3;            // 0..127
    aCol[j] = (c & 7) * 4;       // 0..28
  }
#pragma unroll
  for (int j = 0; j < 2; ++j) {
    const int c = tid + j * 256;
    bRow[j] = c >> 4;            // 0..31
    bCol[j] = (c & 15) * 4;      // 0..60
  }

  v8f acc[4];
#pragma unroll
  for (int nt = 0; nt < 4; ++nt)
#pragma unroll
    for (int r = 0; r < 8; ++r) acc[nt][r] = 0.0f;

  f32x4 ra[4], rb[2];
  // Prologue: issue tile 0 loads.
#pragma unroll
  for (int j = 0; j < 4; ++j)
    ra[j] = *(const f32x4*)&X[(size_t)(m0 + aRow[j]) * DM + aCol[j]];
#pragma unroll
  for (int j = 0; j < 2; ++j)
    rb[j] = *(const f32x4*)&W[(size_t)bRow[j] * DM + n0 + bCol[j]];

  for (int ks = 0; ks < DM / 32; ++ks) {
    __syncthreads();   // previous iteration's LDS reads complete
    // Commit staged registers (fp32 -> bf16) to LDS.
#pragma unroll
    for (int j = 0; j < 4; ++j) {
      bf16x4 h4 = {(bf16)ra[j].x, (bf16)ra[j].y, (bf16)ra[j].z, (bf16)ra[j].w};
      *(bf16x4*)&Al[aRow[j] * LDA + aCol[j]] = h4;
    }
#pragma unroll
    for (int j = 0; j < 2; ++j) {
      bf16x4 h4 = {(bf16)rb[j].x, (bf16)rb[j].y, (bf16)rb[j].z, (bf16)rb[j].w};
      *(bf16x4*)&Bl[bRow[j] * LDB + bCol[j]] = h4;
    }
    __syncthreads();

    // Issue next tile's global loads; they complete during the WMMAs below.
    if (ks < DM / 32 - 1) {
      const int k1 = (ks + 1) * 32;
#pragma unroll
      for (int j = 0; j < 4; ++j)
        ra[j] = *(const f32x4*)&X[(size_t)(m0 + aRow[j]) * DM + k1 + aCol[j]];
#pragma unroll
      for (int j = 0; j < 2; ++j)
        rb[j] = *(const f32x4*)&W[(size_t)(k1 + bRow[j]) * DM + n0 + bCol[j]];
    }

    const v16bf a = frag_a(&Al[wave * 16 * LDA], LDA, lane);
#pragma unroll
    for (int nt = 0; nt < 4; ++nt) {
      const v16bf b = frag_b(&Bl[nt * 16], LDB, lane);
      acc[nt] = wmma_bf16(a, b, acc[nt]);
    }
  }

  // Epilogue: bias add, bf16 store to head-split workspace.
#pragma unroll
  for (int nt = 0; nt < 4; ++nt) {
    const int col = n0 + nt * 16 + l;     // 0..1023
    const int h = col >> 6, d = col & 63;
    const float bias = Bs[col];
#pragma unroll
    for (int r = 0; r < 8; ++r) {
      const int m  = m0 + wave * 16 + hi * 8 + r;      // 0..4095
      const int bb = m >> 11;                           // m / S
      const int s  = m & (S_DIM - 1);
      const int bh = bb * H_DIM + h;
      const float val = acc[nt][r] + bias;
      const size_t idx = (z == 1)
          ? (size_t)(bh * DH + d) * S_DIM + s           // K^T: [bh][d][s]
          : ((size_t)bh * S_DIM + s) * DH + d;          // Q,V: [bh][s][d]
      Out[idx] = (bf16)val;
    }
  }
}

// ---------------------------------------------------------------------------
// Kernel 2: flash attention.  grid = (S/64, B*H), block = 128 (4 waves).
// K/V tiles double-buffered in LDS and filled via async DMA
// (global_load_async_to_lds_b128), overlapping the current block's WMMAs.
// ---------------------------------------------------------------------------
__global__ __launch_bounds__(128) void attn_kernel(
    const bf16* __restrict__ Qh, const bf16* __restrict__ Kt,
    const bf16* __restrict__ Vh, const int* __restrict__ v_mask,
    const int* __restrict__ q_mask, float* __restrict__ out) {
  constexpr int LD = 72;            // 144B rows: 16B aligned, conflict-free
  constexpr unsigned TILE_BYTES = 64 * LD * sizeof(bf16);   // 9216
  __shared__ bf16 Ql[64 * LD];
  __shared__ bf16 Kl[2][64 * LD];   // [d][kv]  (K^T tile), double buffered
  __shared__ bf16 Vl[2][64 * LD];   // [kv][d], double buffered
  __shared__ bf16 Pl[4 * 16 * LD];  // per-wave P transpose staging

  const int tid  = threadIdx.x;
  const int wave = tid >> 5, lane = tid & 31;
  const int l = lane & 15, hi = (lane >> 4) & 1;
  const int q0 = blockIdx.x * 64;
  const int bh = blockIdx.y;
  const int b  = bh >> 4;
  const int h  = bh & 15;

  const bf16* Qg = Qh + (size_t)bh * S_DIM * DH + (size_t)q0 * DH;
  const bf16* Kg = Kt + (size_t)bh * DH * S_DIM;     // [d][s]
  const bf16* Vg = Vh + (size_t)bh * S_DIM * DH;     // [s][d]

  // Stage Q tile (64x64 bf16) once via vector copies.
#pragma unroll
  for (int j = 0; j < 4; ++j) {
    const int c = tid + j * 128;
    const int row = c >> 3, d8 = (c & 7) * 8;
    *(bf16x8*)&Ql[row * LD + d8] = *(const bf16x8*)&Qg[row * DH + d8];
  }
  __syncthreads();
  const v16bf qa0 = frag_a(&Ql[wave * 16 * LD + 0], LD, lane);
  const v16bf qa1 = frag_a(&Ql[wave * 16 * LD + 32], LD, lane);

  // Per-thread async-copy geometry: 4 K chunks + 4 V chunks per tile.
  unsigned kOff[4], vOff[4];
  const bf16* kPtr[4];
  const bf16* vPtr[4];
#pragma unroll
  for (int j = 0; j < 4; ++j) {
    const int c = tid + j * 128;
    const int row = c >> 3, x8 = (c & 7) * 8;
    kOff[j] = lds_off(&Kl[0][row * LD + x8]);
    vOff[j] = lds_off(&Vl[0][row * LD + x8]);
    kPtr[j] = Kg + (size_t)row * S_DIM + x8;   // + kv0       per tile
    vPtr[j] = Vg + (size_t)row * DH + x8;      // + kv0*DH    per tile
  }

  float mrow[8], lrow[8];
  v8f oacc[4];
#pragma unroll
  for (int r = 0; r < 8; ++r) { mrow[r] = -3.0e38f; lrow[r] = 0.0f; }
#pragma unroll
  for (int nt = 0; nt < 4; ++nt)
#pragma unroll
    for (int r = 0; r < 8; ++r) oacc[nt][r] = 0.0f;

  bf16* Pw = &Pl[wave * 16 * LD];

  // Prologue: async-load kv block 0 into buffer 0.
#pragma unroll
  for (int j = 0; j < 4; ++j) {
    async_copy_b128(kOff[j], kPtr[j]);
    async_copy_b128(vOff[j], vPtr[j]);
  }

  for (int kb = 0; kb < S_DIM / 64; ++kb) {
    wait_asynccnt0();    // my DMA ops for block kb have landed in LDS
    __syncthreads();     // everyone's landed; everyone done reading buf !cur

    // Kick off next kv block's DMA into the other buffer (overlaps WMMAs).
    if (kb < S_DIM / 64 - 1) {
      const unsigned bsel = (unsigned)((kb + 1) & 1) * TILE_BYTES;
      const int kv1 = (kb + 1) * 64;
#pragma unroll
      for (int j = 0; j < 4; ++j) {
        async_copy_b128(kOff[j] + bsel, kPtr[j] + kv1);
        async_copy_b128(vOff[j] + bsel, vPtr[j] + (size_t)kv1 * DH);
      }
    }

    const bf16* Kc = &Kl[kb & 1][0];
    const bf16* Vc = &Vl[kb & 1][0];
    const int kv0 = kb * 64;

    // S = Q K^T : A = Q rows, B = K^T ([k=d][n=kv] row-major in Kc)
    v8f sacc[4];
#pragma unroll
    for (int nt = 0; nt < 4; ++nt) {
#pragma unroll
      for (int r = 0; r < 8; ++r) sacc[nt][r] = 0.0f;
      const v16bf kf0 = frag_b(&Kc[0 * LD + nt * 16], LD, lane);
      sacc[nt] = wmma_bf16(qa0, kf0, sacc[nt]);
      const v16bf kf1 = frag_b(&Kc[32 * LD + nt * 16], LD, lane);
      sacc[nt] = wmma_bf16(qa1, kf1, sacc[nt]);
    }

    // scale + additive kv mask (lane's N is the kv offset)
    float madd[4];
#pragma unroll
    for (int nt = 0; nt < 4; ++nt) {
      const int kv = kv0 + nt * 16 + l;
      madd[nt] = (1.0f - (float)v_mask[b * S_DIM + kv]) * -1.0e10f;
    }
#pragma unroll
    for (int nt = 0; nt < 4; ++nt)
#pragma unroll
      for (int r = 0; r < 8; ++r)
        sacc[nt][r] = sacc[nt][r] * 0.125f + madd[nt];

    // online softmax: row lives across 16 lanes in VGPR slot r
    float alpha[8];
#pragma unroll
    for (int r = 0; r < 8; ++r) {
      float c = fmaxf(fmaxf(sacc[0][r], sacc[1][r]),
                      fmaxf(sacc[2][r], sacc[3][r]));
      c = fmaxf(c, __shfl_xor(c, 1, 32));
      c = fmaxf(c, __shfl_xor(c, 2, 32));
      c = fmaxf(c, __shfl_xor(c, 4, 32));
      c = fmaxf(c, __shfl_xor(c, 8, 32));
      const float mn = fmaxf(mrow[r], c);
      alpha[r] = __expf(mrow[r] - mn);
      mrow[r]  = mn;
    }
    float psum[8];
#pragma unroll
    for (int r = 0; r < 8; ++r) psum[r] = 0.0f;
#pragma unroll
    for (int nt = 0; nt < 4; ++nt)
#pragma unroll
      for (int r = 0; r < 8; ++r) {
        const float p = __expf(sacc[nt][r] - mrow[r]);
        psum[r] += p;
        // transpose accum layout -> A layout via wave-private LDS patch
        Pw[(hi * 8 + r) * LD + nt * 16 + l] = (bf16)p;
      }
#pragma unroll
    for (int r = 0; r < 8; ++r) {
      float ps = psum[r];
      ps += __shfl_xor(ps, 1, 32);
      ps += __shfl_xor(ps, 2, 32);
      ps += __shfl_xor(ps, 4, 32);
      ps += __shfl_xor(ps, 8, 32);
      lrow[r] = lrow[r] * alpha[r] + ps;
    }
#pragma unroll
    for (int nt = 0; nt < 4; ++nt)
#pragma unroll
      for (int r = 0; r < 8; ++r) oacc[nt][r] *= alpha[r];

    // O += P V : A = P (16x64), B = V ([k=kv][n=d] row-major in Vc)
    const v16bf pa0 = frag_a(Pw, LD, lane);
    const v16bf pa1 = frag_a(Pw + 32, LD, lane);
#pragma unroll
    for (int nt = 0; nt < 4; ++nt) {
      const v16bf vf0 = frag_b(&Vc[0 * LD + nt * 16], LD, lane);
      oacc[nt] = wmma_bf16(pa0, vf0, oacc[nt]);
      const v16bf vf1 = frag_b(&Vc[32 * LD + nt * 16], LD, lane);
      oacc[nt] = wmma_bf16(pa1, vf1, oacc[nt]);
    }
  }

  // Epilogue: normalize, apply q mask, fp32 store [b][s][h*64+d]
#pragma unroll
  for (int r = 0; r < 8; ++r) {
    const int qrow = q0 + wave * 16 + hi * 8 + r;
    const float qm = (float)q_mask[b * S_DIM + qrow];
    const float rinv = qm / lrow[r];
#pragma unroll
    for (int nt = 0; nt < 4; ++nt) {
      out[((size_t)(b * S_DIM + qrow)) * (H_DIM * DH) + h * DH + nt * 16 + l] =
          oacc[nt][r] * rinv;
    }
  }
}

// ---------------------------------------------------------------------------
extern "C" void kernel_launch(void* const* d_in, const int* in_sizes, int n_in,
                              void* d_out, int out_size, void* d_ws,
                              size_t ws_size, hipStream_t stream) {
  (void)in_sizes; (void)n_in; (void)out_size; (void)ws_size;
  const float* q      = (const float*)d_in[0];
  const float* k      = (const float*)d_in[1];
  const float* v      = (const float*)d_in[2];
  const int*   v_mask = (const int*)d_in[3];
  const int*   q_mask = (const int*)d_in[4];
  const float* Wq     = (const float*)d_in[5];
  const float* bq     = (const float*)d_in[6];
  const float* Wk     = (const float*)d_in[7];
  const float* bk     = (const float*)d_in[8];
  const float* Wv     = (const float*)d_in[9];
  const float* bv     = (const float*)d_in[10];
  float* out = (float*)d_out;

  bf16* Qh = (bf16*)d_ws;                 // 8 MB  [bh][s][d]
  bf16* Kh = Qh + PROJ_ELEMS;             // 8 MB  [bh][d][s]  (transposed)
  bf16* Vh = Kh + PROJ_ELEMS;             // 8 MB  [bh][s][d]

  dim3 g1(DM / 64, (B_DIM * S_DIM) / 128, 3);   // (16, 32, 3)
  qkv_proj_kernel<<<g1, 256, 0, stream>>>(q, k, v, Wq, Wk, Wv, bq, bk, bv,
                                          Qh, Kh, Vh);

  dim3 g2(S_DIM / 64, BH);                      // (32, 32)
  attn_kernel<<<g2, 128, 0, stream>>>(Qh, Kh, Vh, v_mask, q_mask, out);
}
